// PraxisMemory_73933567033857
// MI455X (gfx1250) — matrix operand order repaired and checked by reading, not compiled, not run on hardware
//
#include <hip/hip_runtime.h>
#include <hip/hip_bf16.h>

#define EPSF 1e-8f

// ---- CDNA5 WMMA types (wave32) ----
typedef __attribute__((ext_vector_type(16))) __bf16 v16bf;
typedef __attribute__((ext_vector_type(8)))  __bf16 v8bf;
typedef __attribute__((ext_vector_type(4)))  __bf16 v4bf;
typedef __attribute__((ext_vector_type(2)))  __bf16 v2bf;
typedef __attribute__((ext_vector_type(8)))  float  v8f;

union V16 { v16bf v; v8bf h[2]; };

__device__ __forceinline__ v8f wmma_bf16(v16bf a, v16bf b, v8f c) {
  // D = A(16x32 bf16) * B(32x16 bf16) + C(16x16 f32)
  return __builtin_amdgcn_wmma_f32_16x16x32_bf16(
      /*neg_a=*/false, a, /*neg_b=*/false, b,
      /*c_mod=*/(short)0, c, /*reuse_a=*/false, /*reuse_b=*/false);
}

__device__ __forceinline__ float elu1(float x) {
  return x > 0.f ? x + 1.f : __expf(x);   // elu(x)+1
}

__device__ __forceinline__ float fast_rcp(float x) {
  return __builtin_amdgcn_rcpf(x);        // v_rcp_f32 (den > 0, approx is fine)
}

__device__ __forceinline__ void lds_fence() {
  asm volatile("s_wait_dscnt 0" ::: "memory");
}

constexpr int NB = 2, NH = 16, NS = 4096, ND = 64, NBH = NB * NH;

// ---------------------------------------------------------------------------
// Kernel 0: workspace init.  Sbuf[b,h] = 2*init_states[h]; Zbuf[b,h] = init_z[h]
// ---------------------------------------------------------------------------
__global__ void pm_init(const float* __restrict__ init_states,
                        const float* __restrict__ init_z,
                        float* __restrict__ Sbuf, float* __restrict__ Zbuf) {
  int idx = blockIdx.x * 256 + threadIdx.x;        // 0 .. NBH*64*64-1
  int h = (idx >> 12) & (NH - 1);
  Sbuf[idx] = 2.f * init_states[(h << 12) + (idx & 4095)];
  if (idx < NBH * ND) {
    int hz = (idx >> 6) & (NH - 1);
    Zbuf[idx] = init_z[(hz << 6) + (idx & 63)];
  }
}

// ---------------------------------------------------------------------------
// Kernel 1: key/value pass.
//   sigma_k = elu(K)+1 ; ret = sigma_k @ M0 ; den = sigma_k . z0 + eps
//   delta = V - ret/den ; Sbuf += sigma_k^T @ delta ; Zbuf += col-sums(sigma_k)
// grid = NBH*8 blocks of 128 threads (4 waves); each wave: 4 chunks of 32 rows
// ---------------------------------------------------------------------------
__global__ void __launch_bounds__(128) pm_pass1(
    const float* __restrict__ key, const float* __restrict__ value,
    const float* __restrict__ init_states, const float* __restrict__ init_z,
    float* __restrict__ Sbuf, float* __restrict__ Zbuf) {
  const int bh = blockIdx.x >> 3;
  const int g  = blockIdx.x & 7;
  const int h  = bh & (NH - 1);
  const int tid = threadIdx.x;
  const int wv = tid >> 5;          // wave in block, 0..3
  const int L  = tid & 31;          // lane
  const int m  = L & 15;
  const int kh = L >> 4;

  __shared__ alignas(16) __bf16 sinitT[64][72];     // sinitT[e][d] = M0[d][e]
  __shared__ float zin[64];
  __shared__ alignas(16) __bf16 sig[4][32][72];     // sigma_k tile, row-major
  __shared__ alignas(16) __bf16 dltT[4][64][40];    // delta^T: [e][s]

  for (int i = tid; i < 4096; i += 128) {
    int d = i >> 6, e = i & 63;
    sinitT[e][d] = (__bf16)init_states[(h << 12) + i];
  }
  if (tid < 64) zin[tid] = init_z[(h << 6) + tid];
  __syncthreads();

  const v8f vzero = {0.f,0.f,0.f,0.f,0.f,0.f,0.f,0.f};
  v8f acc[16];
  #pragma unroll
  for (int i = 0; i < 16; ++i) acc[i] = vzero;
  float zacc0 = 0.f, zacc1 = 0.f;
  const int gw = g * 4 + wv;        // 0..31 wave id within (b,h)

  #pragma unroll 1
  for (int cc = 0; cc < 4; ++cc) {
    const int s0 = (gw + cc * 32) * 32;

    // prefetch next chunk's K/V rows into L2 while we compute this chunk
    if (cc < 3) {
      const int sn = (gw + (cc + 1) * 32) * 32;
      const float* kn = key   + ((size_t)bh * NS + sn + L) * ND;
      const float* vn = value + ((size_t)bh * NS + sn + L) * ND;
      __builtin_prefetch(kn, 0, 0);
      __builtin_prefetch(kn + 32, 0, 0);
      __builtin_prefetch(vn, 0, 0);
      __builtin_prefetch(vn + 32, 0, 0);
    }

    // -------- stage 1: sigma_k -> LDS (bf16), per-row denominator --------
    const float* krow = key + ((size_t)bh * NS + s0 + L) * ND;  // lane owns row L
    float den = EPSF;
    #pragma unroll
    for (int j = 0; j < 16; ++j) {
      float4 q = ((const float4*)krow)[j];
      float a0 = elu1(q.x), a1 = elu1(q.y), a2 = elu1(q.z), a3 = elu1(q.w);
      den += a0 * zin[4*j+0] + a1 * zin[4*j+1] + a2 * zin[4*j+2] + a3 * zin[4*j+3];
      v4bf p; p[0] = (__bf16)a0; p[1] = (__bf16)a1; p[2] = (__bf16)a2; p[3] = (__bf16)a3;
      *(v4bf*)&sig[wv][L][4*j] = p;
    }
    const float rden = fast_rcp(den);   // one v_rcp_f32 per row, broadcast later
    lds_fence();

    // -------- z column partial sums (lane L -> cols 2L, 2L+1: one b32/row) ---
    #pragma unroll 8
    for (int s = 0; s < 32; ++s) {
      v2bf pp = *(const v2bf*)&sig[wv][s][2 * L];
      zacc0 += (float)pp[0];
      zacc1 += (float)pp[1];
    }

    // -------- stage 2: ret = sigma @ M0 ; delta -> dltT --------
    #pragma unroll
    for (int t = 0; t < 2; ++t) {
      const int rowL = 16 * t + m;                 // A row = lane%16
      V16 A0, A1;                                  // K in {8kh..+8, 16+8kh..+8}
      A0.h[0] = *(const v8bf*)&sig[wv][rowL][ 8*kh];
      A0.h[1] = *(const v8bf*)&sig[wv][rowL][16 + 8*kh];
      A1.h[0] = *(const v8bf*)&sig[wv][rowL][32 + 8*kh];
      A1.h[1] = *(const v8bf*)&sig[wv][rowL][48 + 8*kh];
      float rdnm[8];
      #pragma unroll
      for (int r = 0; r < 8; ++r) rdnm[r] = __shfl(rden, 16*t + 8*kh + r, 32);
      #pragma unroll
      for (int u = 0; u < 4; ++u) {
        const int n = 16 * u + m;                  // B col = lane%16
        V16 B0, B1;                                // B[k][n] = sinitT[n][k]
        B0.h[0] = *(const v8bf*)&sinitT[n][8*kh];
        B0.h[1] = *(const v8bf*)&sinitT[n][16 + 8*kh];
        B1.h[0] = *(const v8bf*)&sinitT[n][32 + 8*kh];
        B1.h[1] = *(const v8bf*)&sinitT[n][48 + 8*kh];
        v8f c8 = vzero;
        c8 = wmma_bf16(A0.v, B0.v, c8);
        c8 = wmma_bf16(A1.v, B1.v, c8);
        // delta = V - ret*rcp(den), stored transposed: dltT[e][s]
        const float* vcol = value + ((size_t)bh * NS + s0) * ND + n;
        v8bf dv;
        #pragma unroll
        for (int r = 0; r < 8; ++r) {
          const int rr = 16*t + 8*kh + r;          // C row M = r + 8*kh
          dv[r] = (__bf16)(vcol[rr * ND] - c8[r] * rdnm[r]);
        }
        *(v8bf*)&dltT[wv][n][16*t + 8*kh] = dv;
      }
    }
    lds_fence();

    // -------- stage 3: acc += sigma^T @ delta (K = 32 rows of s) --------
    #pragma unroll
    for (int dt = 0; dt < 4; ++dt) {
      const int d = 16 * dt + m;                   // A' row = feature dim d
      V16 Ap;
      #pragma unroll
      for (int i = 0; i < 8; ++i) Ap.v[i]     = sig[wv][ 8*kh + i][d];
      #pragma unroll
      for (int i = 0; i < 8; ++i) Ap.v[8 + i] = sig[wv][16 + 8*kh + i][d];
      #pragma unroll
      for (int et = 0; et < 4; ++et) {
        const int n = 16 * et + m;
        V16 Bp;                                    // B'[k=s][n=e] = dltT[n][s]
        Bp.h[0] = *(const v8bf*)&dltT[wv][n][8*kh];
        Bp.h[1] = *(const v8bf*)&dltT[wv][n][16 + 8*kh];
        acc[dt*4 + et] = wmma_bf16(Ap.v, Bp.v, acc[dt*4 + et]);
      }
    }
  }

  // -------- flush partial accumulators --------
  float* sb = Sbuf + (size_t)bh * 4096;
  #pragma unroll
  for (int dt = 0; dt < 4; ++dt)
    #pragma unroll
    for (int et = 0; et < 4; ++et)
      #pragma unroll
      for (int r = 0; r < 8; ++r) {
        int dd = 16*dt + 8*kh + r;
        int ee = 16*et + m;
        atomicAdd(&sb[dd * 64 + ee], acc[dt*4 + et][r]);
      }
  atomicAdd(&Zbuf[bh * 64 + 2*L],     zacc0);
  atomicAdd(&Zbuf[bh * 64 + 2*L + 1], zacc1);
}

// ---------------------------------------------------------------------------
// Kernel 2: query pass.
//   sigma_q = elu(Q)+1 ; retr = sigma_q @ M ; norm = sigma_q . mz + eps
//   out = gate * retr/norm + (1-gate) * attn_out
// grid = NBH*8 blocks of 256 threads (8 waves); each wave: 4 tiles of 16 rows
// ---------------------------------------------------------------------------
__global__ void __launch_bounds__(256) pm_pass2(
    const float* __restrict__ query, const float* __restrict__ attn,
    const float* __restrict__ betas,
    const float* __restrict__ Sbuf, const float* __restrict__ Zbuf,
    float* __restrict__ out) {
  const int bh = blockIdx.x >> 3;
  const int g  = blockIdx.x & 7;
  const int h  = bh & (NH - 1);
  const int tid = threadIdx.x;
  const int wv = tid >> 5;          // 0..7
  const int L  = tid & 31;
  const int m  = L & 15;
  const int kh = L >> 4;

  __shared__ alignas(16) __bf16 stT[64][72];       // stT[e][d] = M[d][e]
  __shared__ float mz[64];
  __shared__ float gt[64];

  for (int i = tid; i < 4096; i += 256) {
    int d = i >> 6, e = i & 63;
    stT[e][d] = (__bf16)Sbuf[((size_t)bh << 12) + i];
  }
  if (tid < 64) {
    mz[tid] = Zbuf[bh * 64 + tid];
    gt[tid] = 1.f / (1.f + __expf(-betas[h * 64 + tid]));   // sigmoid(beta)
  }
  __syncthreads();

  const v8f vzero = {0.f,0.f,0.f,0.f,0.f,0.f,0.f,0.f};
  const int gw = g * 8 + wv;        // 0..63

  #pragma unroll 1
  for (int cc = 0; cc < 4; ++cc) {
    const int s0 = (gw + cc * 64) * 16;

    // prefetch next tile's Q / attn rows
    if (cc < 3) {
      const int sn = (gw + (cc + 1) * 64) * 16;
      __builtin_prefetch(query + ((size_t)bh * NS + sn + m) * ND, 0, 0);
      __builtin_prefetch(attn  + ((size_t)bh * NS + sn + m) * ND, 0, 0);
    }

    const float* qrow = query + ((size_t)bh * NS + s0 + m) * ND;
    float nrm = 0.f;
    V16 A[2];
    #pragma unroll
    for (int c = 0; c < 2; ++c) {
      const int b1 = 32*c + 8*kh;
      const int b2 = b1 + 16;
      float4 x0 = *(const float4*)(qrow + b1);
      float4 x1 = *(const float4*)(qrow + b1 + 4);
      float4 y0 = *(const float4*)(qrow + b2);
      float4 y1 = *(const float4*)(qrow + b2 + 4);
      float f[16] = {x0.x,x0.y,x0.z,x0.w, x1.x,x1.y,x1.z,x1.w,
                     y0.x,y0.y,y0.z,y0.w, y1.x,y1.y,y1.z,y1.w};
      #pragma unroll
      for (int i = 0; i < 16; ++i) {
        int k = (i < 8) ? (b1 + i) : (b2 + i - 8);
        float sv = elu1(f[i]);
        nrm += sv * mz[k];
        A[c].v[i] = (__bf16)sv;
      }
    }
    nrm += __shfl_xor(nrm, 16, 32);  // combine the two K-halves of row m
    const float rnrm = fast_rcp(nrm + EPSF);   // one v_rcp_f32 per row
    #pragma unroll
    for (int u = 0; u < 4; ++u) {
      const int n = 16 * u + m;
      V16 B0, B1;
      B0.h[0] = *(const v8bf*)&stT[n][8*kh];
      B0.h[1] = *(const v8bf*)&stT[n][16 + 8*kh];
      B1.h[0] = *(const v8bf*)&stT[n][32 + 8*kh];
      B1.h[1] = *(const v8bf*)&stT[n][48 + 8*kh];
      v8f c8 = vzero;
      c8 = wmma_bf16(A[0].v, B0.v, c8);
      c8 = wmma_bf16(A[1].v, B1.v, c8);
      const float gg = gt[n];
      const float* acol = attn + ((size_t)bh * NS + s0) * ND + n;
      float*       ocol = out  + ((size_t)bh * NS + s0) * ND + n;
      #pragma unroll
      for (int r = 0; r < 8; ++r) {
        const int rr = 8*kh + r;                   // C row M
        float nr = __shfl(rnrm, rr, 32);
        float mo = c8[r] * nr;
        ocol[(size_t)rr * ND] = gg * mo + (1.f - gg) * acol[(size_t)rr * ND];
      }
    }
  }
}

// ---------------------------------------------------------------------------
extern "C" void kernel_launch(void* const* d_in, const int* in_sizes, int n_in,
                              void* d_out, int out_size, void* d_ws, size_t ws_size,
                              hipStream_t stream) {
  (void)in_sizes; (void)n_in; (void)out_size; (void)ws_size;
  const float* query = (const float*)d_in[0];
  const float* key   = (const float*)d_in[1];
  const float* value = (const float*)d_in[2];
  const float* attn  = (const float*)d_in[3];
  const float* betas = (const float*)d_in[4];
  const float* inits = (const float*)d_in[5];
  const float* initz = (const float*)d_in[6];
  float* out  = (float*)d_out;
  float* Sbuf = (float*)d_ws;                 // [NBH][64][64] f32 accumulators
  float* Zbuf = Sbuf + (size_t)NBH * 64 * 64; // [NBH][64]

  pm_init <<<dim3(NBH * 64 * 64 / 256), dim3(256), 0, stream>>>(inits, initz, Sbuf, Zbuf);
  pm_pass1<<<dim3(NBH * 8),             dim3(128), 0, stream>>>(key, value, inits, initz, Sbuf, Zbuf);
  pm_pass2<<<dim3(NBH * 8),             dim3(256), 0, stream>>>(query, attn, betas, Sbuf, Zbuf, out);
}
